// LightweightIndustrialDiffusion_8297876816595
// MI455X (gfx1250) — compile-verified, matrix operand order checked by reading, not run on hardware
//
#include <hip/hip_runtime.h>
#include <hip/hip_bf16.h>
#include <math.h>

typedef __attribute__((ext_vector_type(16))) _Float16 v16h;
typedef __attribute__((ext_vector_type(8)))  _Float16 v8h;
typedef __attribute__((ext_vector_type(8)))  float    v8f;

#define NN   16384      // nodes
#define EE   131072     // edges
#define BB   256        // graphs
#define NPG  64
#define HID  64
#define NH   4

// ---------------- float <-> monotonic uint key (for atomic max on float) ----
__device__ __forceinline__ unsigned f2key(float f) {
  unsigned b = __float_as_uint(f);
  return (b & 0x80000000u) ? ~b : (b | 0x80000000u);
}
__device__ __forceinline__ float key2f(unsigned k) {
  unsigned b = (k & 0x80000000u) ? (k ^ 0x80000000u) : ~k;
  return __uint_as_float(b);
}
#define NEG_INF_KEY 0x007FFFFFu   // key of -inf

// ---------------- fills -----------------------------------------------------
__global__ void k_fill_f32(float* p, int n, float v) {
  int i = blockIdx.x * blockDim.x + threadIdx.x;
  if (i < n) p[i] = v;
}
__global__ void k_fill_u32(unsigned* p, int n, unsigned v) {
  int i = blockIdx.x * blockDim.x + threadIdx.x;
  if (i < n) p[i] = v;
}
__global__ void k_f32_to_f16(const float* __restrict__ s, _Float16* __restrict__ d, int n) {
  int i = blockIdx.x * blockDim.x + threadIdx.x;
  if (i < n) d[i] = (_Float16)s[i];
}

// ---------------- pack B weight [64,Nc] f32 -> WMMA lane-order f16 ----------
// Layout: dst[ ((tc*2+ks)*32 + lane)*16 + i ] so each lane's 16-half fragment
// is contiguous (two aligned b128 loads in the GEMM).
__global__ void k_pack_b(const float* __restrict__ src, _Float16* __restrict__ dst, int Nc) {
  int o = blockIdx.x * blockDim.x + threadIdx.x;
  if (o >= 64 * Nc) return;
  int i    = o & 15;
  int lane = (o >> 4) & 31;
  int ks   = (o >> 9) & 1;
  int tc   = o >> 10;
  int kb   = (lane >> 4) << 3;
  int j = i >> 1, p = i & 1;
  int kk = (j < 4) ? (kb + 2 * j + p) : (16 + kb + 2 * (j - 4) + p);
  int k  = ks * 32 + kk;
  int col = tc * 16 + (lane & 15);
  dst[o] = (_Float16)src[k * Nc + col];
}

// ---------------- prep: time embedding + constant column folds --------------
__global__ void k_prep(const int* tptr, const float* W_time, const float* b_time,
                       const float* Wq1, const float* bq1,
                       const float* Wk1, const float* bk1,
                       const float* Wv1, const float* bv1,
                       const float* Ws1, const float* bs1,
                       float* cq1, float* ck1, float* cv1, float* cs1) {
  __shared__ float semb[16], stemb[16];
  int tid = threadIdx.x;
  if (tid < 8) {
    float scale = logf(10000.0f) / 7.0f;
    float freq  = expf(-(float)tid * scale);
    float arg   = ((float)tptr[0] / 100.0f) * freq;
    semb[tid]     = sinf(arg);
    semb[tid + 8] = cosf(arg);
  }
  __syncthreads();
  if (tid < 16) {
    float s = b_time[tid];
    for (int j = 0; j < 16; ++j) s += semb[j] * W_time[j * 16 + tid];
    stemb[tid] = s;
  }
  __syncthreads();
  if (tid < 256) {
    float sq = bq1[tid], sk = bk1[tid], sv = bv1[tid];
    for (int j = 0; j < 16; ++j) {
      float tv = stemb[j];
      sq += tv * Wq1[(4 + j) * 256 + tid];
      sk += tv * Wk1[(4 + j) * 256 + tid];
      sv += tv * Wv1[(4 + j) * 256 + tid];
    }
    cq1[tid] = sq; ck1[tid] = sk; cv1[tid] = sv;
    if (tid < 64) {
      float ss = bs1[tid];
      for (int j = 0; j < 16; ++j) ss += stemb[j] * Ws1[(4 + j) * 64 + tid];
      cs1[tid] = ss;
    }
  }
}

// ---------------- layer-1 projections (x has only 4 cols) -------------------
__global__ void k_proj1(const float* __restrict__ x,
                        const float* __restrict__ Wq1, const float* __restrict__ Wk1,
                        const float* __restrict__ Wv1,
                        const float* __restrict__ cq1, const float* __restrict__ ck1,
                        const float* __restrict__ cv1,
                        float* __restrict__ q, float* __restrict__ k, float* __restrict__ v) {
  int idx = blockIdx.x * blockDim.x + threadIdx.x;
  if (idx >= NN * 256) return;
  int n = idx >> 8, c = idx & 255;
  float x0 = x[n * 4 + 0], x1 = x[n * 4 + 1], x2 = x[n * 4 + 2], x3 = x[n * 4 + 3];
  q[idx] = cq1[c] + x0 * Wq1[c] + x1 * Wq1[256 + c] + x2 * Wq1[512 + c] + x3 * Wq1[768 + c];
  k[idx] = ck1[c] + x0 * Wk1[c] + x1 * Wk1[256 + c] + x2 * Wk1[512 + c] + x3 * Wk1[768 + c];
  v[idx] = cv1[c] + x0 * Wv1[c] + x1 * Wv1[256 + c] + x2 * Wv1[512 + c] + x3 * Wv1[768 + c];
}
__global__ void k_skip1(const float* __restrict__ x, const float* __restrict__ Ws1,
                        const float* __restrict__ cs1, float* __restrict__ skip) {
  int idx = blockIdx.x * blockDim.x + threadIdx.x;
  if (idx >= NN * 64) return;
  int n = idx >> 6, c = idx & 63;
  skip[idx] = cs1[c] + x[n * 4 + 0] * Ws1[c] + x[n * 4 + 1] * Ws1[64 + c] +
              x[n * 4 + 2] * Ws1[128 + c] + x[n * 4 + 3] * Ws1[192 + c];
}

// ---------------- attention: logits + segment max ---------------------------
__global__ void k_logits(const float* __restrict__ q, const float* __restrict__ k,
                         const int* __restrict__ ei, float* __restrict__ logits,
                         unsigned* __restrict__ nmax) {
  int idx = blockIdx.x * blockDim.x + threadIdx.x;
  if (idx >= EE * NH) return;
  int e = idx >> 2, h = idx & 3;
  int s = ei[e], d = ei[EE + e];
  const float* qp = q + (size_t)d * 256 + h * 64;
  const float* kp = k + (size_t)s * 256 + h * 64;
  float acc = 0.f;
  for (int c = 0; c < 64; ++c) acc += qp[c] * kp[c];
  acc *= 0.125f;                       // 1/sqrt(64)
  logits[idx] = acc;
  atomicMax(&nmax[d * NH + h], f2key(acc));
}
// ---------------- attention: exp + segment sum ------------------------------
__global__ void k_expdenom(float* __restrict__ logits, const int* __restrict__ ei,
                           const unsigned* __restrict__ nmax, float* __restrict__ denom) {
  int idx = blockIdx.x * blockDim.x + threadIdx.x;
  if (idx >= EE * NH) return;
  int e = idx >> 2, h = idx & 3;
  int d = ei[EE + e];
  float m  = key2f(nmax[d * NH + h]);
  float ex = expf(logits[idx] - m);
  logits[idx] = ex;                    // store ex in-place
  unsafeAtomicAdd(&denom[d * NH + h], ex);
}
// ---------------- attention: weighted scatter -------------------------------
__global__ void k_scatter(const float* __restrict__ v, const float* __restrict__ ex,
                          const int* __restrict__ ei, const float* __restrict__ denom,
                          float* __restrict__ agg) {
  int idx = blockIdx.x * blockDim.x + threadIdx.x;
  if (idx >= EE * NH) return;
  int e = idx >> 2, h = idx & 3;
  int s = ei[e], d = ei[EE + e];
  float dn = denom[d * NH + h];
  float w  = ex[idx] / (dn == 0.f ? 1.f : dn);
  const float* vp = v + (size_t)s * 256 + h * 64;
  float* ap = agg + (size_t)d * 256 + h * 64;
  for (int c = 0; c < 64; ++c) unsafeAtomicAdd(&ap[c], vp[c] * w);
}
// ---------------- head mean + skip + relu -----------------------------------
__global__ void k_combine(const float* __restrict__ agg, const float* __restrict__ skip,
                          float* __restrict__ h) {
  int idx = blockIdx.x * blockDim.x + threadIdx.x;
  if (idx >= NN * 64) return;
  int n = idx >> 6, c = idx & 63;
  const float* ap = agg + (size_t)n * 256;
  float s = 0.25f * (ap[c] + ap[64 + c] + ap[128 + c] + ap[192 + c]) + skip[idx];
  h[idx] = s > 0.f ? s : 0.f;
}

// ---------------- WMMA GEMM: C[M,NC] = A[M,64](f16) @ Bpk(f16,packed) + bias
// One 16x16 tile per wave; K=64 -> two v_wmma_f32_16x16x32_f16.
// A fragment  : two contiguous 8-half runs per lane -> b128 loads.
// B fragment  : pre-swizzled by k_pack_b -> one contiguous 16-half run.
template <int NC>
__global__ void k_gemm_wmma(const _Float16* __restrict__ A, const _Float16* __restrict__ Bpk,
                            const float* __restrict__ bias, float* __restrict__ C, int M) {
  constexpr int TPR = NC >> 4;
  int wave = threadIdx.x >> 5;
  int lane = threadIdx.x & 31;
  int totalTiles = (M >> 4) * TPR;
  int tile = blockIdx.x * (blockDim.x >> 5) + wave;
  if (tile >= totalTiles) return;          // wave-uniform: EXEC all-ones at WMMA
  int tr = tile / TPR, tc = tile % TPR;
  int row0 = tr << 4;
  int m  = lane & 15;
  int kb = (lane >> 4) << 3;               // 0 or 8

  const _Float16* Arow = A + (size_t)(row0 + m) * 64;
  v8h aL0 = *(const v8h*)(Arow + kb);
  v8h aH0 = *(const v8h*)(Arow + 16 + kb);
  v8h aL1 = *(const v8h*)(Arow + 32 + kb);
  v8h aH1 = *(const v8h*)(Arow + 48 + kb);

  const _Float16* bp = Bpk + (size_t)tc * 1024 + lane * 16;
  v8h bL0 = *(const v8h*)(bp);
  v8h bH0 = *(const v8h*)(bp + 8);
  v8h bL1 = *(const v8h*)(bp + 512);
  v8h bH1 = *(const v8h*)(bp + 520);

  v16h a0 = __builtin_shufflevector(aL0, aH0, 0,1,2,3,4,5,6,7,8,9,10,11,12,13,14,15);
  v16h a1 = __builtin_shufflevector(aL1, aH1, 0,1,2,3,4,5,6,7,8,9,10,11,12,13,14,15);
  v16h b0 = __builtin_shufflevector(bL0, bH0, 0,1,2,3,4,5,6,7,8,9,10,11,12,13,14,15);
  v16h b1 = __builtin_shufflevector(bL1, bH1, 0,1,2,3,4,5,6,7,8,9,10,11,12,13,14,15);

  v8f c = {};
  c = __builtin_amdgcn_wmma_f32_16x16x32_f16(false, a0, false, b0, (short)0, c, false, false);
  c = __builtin_amdgcn_wmma_f32_16x16x32_f16(false, a1, false, b1, (short)0, c, false, false);

  float bv = bias ? bias[tc * 16 + m] : 0.f;
  float* Crow = C + (size_t)(row0 + kb) * NC + tc * 16 + m;  // lanes 16-31: rows +8
#pragma unroll
  for (int r = 0; r < 8; ++r) Crow[(size_t)r * NC] = c[r] + bv;
}

// ---------------- node logits ----------------------------------------------
__global__ void k_node(const float* __restrict__ h2, const float* __restrict__ Wn,
                       const float* __restrict__ bn, float* __restrict__ out) {
  int idx = blockIdx.x * blockDim.x + threadIdx.x;
  if (idx >= NN * 4) return;
  int n = idx >> 2, c = idx & 3;
  float s = bn[c];
  const float* hp = h2 + (size_t)n * 64;
  for (int k = 0; k < 64; ++k) s += hp[k] * Wn[k * 4 + c];
  out[idx] = s;
}

// ---------------- fused pairwise edge MLP -----------------------------------
// edge_logits[b,i,j,:] = relu(Apre[b,i,:] + Bpre[b,j,:]) @ W_e2 + b_e2
__global__ void k_edge(const float* __restrict__ Apre, const float* __restrict__ Bpre,
                       const float* __restrict__ W_e2, const float* __restrict__ b_e2,
                       float* __restrict__ out) {
  __shared__ float As[16 * 64], Bs[16 * 64], wa[64], wb[64];
  int b  = blockIdx.x >> 2;
  int i0 = (blockIdx.x & 3) << 4;
  int tid = threadIdx.x;
  for (int idx = tid; idx < 1024; idx += 256)
    As[idx] = Apre[(size_t)(b * 64 + i0 + (idx >> 6)) * 64 + (idx & 63)];
  if (tid < 64) { wa[tid] = W_e2[tid * 2]; wb[tid] = W_e2[tid * 2 + 1]; }
  float be0 = b_e2[0], be1 = b_e2[1];
  for (int jt = 0; jt < 4; ++jt) {
    __syncthreads();
    for (int idx = tid; idx < 1024; idx += 256)
      Bs[idx] = Bpre[(size_t)(b * 64 + jt * 16 + (idx >> 6)) * 64 + (idx & 63)];
    if (jt < 3)  // prefetch next j-tile (global_prefetch_b8)
      __builtin_prefetch(&Bpre[(size_t)(b * 64 + (jt + 1) * 16 + (tid >> 6)) * 64 + (tid & 63)], 0, 3);
    __syncthreads();
    int ti = tid >> 4, tj = tid & 15;
    const float* ar = &As[ti * 64];
    const float* br = &Bs[tj * 64];
    float a0 = 0.f, a1 = 0.f;
#pragma unroll 8
    for (int c = 0; c < 64; ++c) {
      float vv = ar[c] + br[c];
      vv = vv > 0.f ? vv : 0.f;
      a0 += vv * wa[c];
      a1 += vv * wb[c];
    }
    size_t base = (size_t)NN * 4 + (((size_t)b * 64 + i0 + ti) * 64 + (jt * 16 + tj)) * 2;
    out[base]     = a0 + be0;
    out[base + 1] = a1 + be1;
  }
}

// ============================================================================
extern "C" void kernel_launch(void* const* d_in, const int* in_sizes, int n_in,
                              void* d_out, int out_size, void* d_ws, size_t ws_size,
                              hipStream_t stream) {
  const float* x       = (const float*)d_in[0];
  const int*   ei      = (const int*)d_in[1];
  const int*   tptr    = (const int*)d_in[3];
  const float* W_time  = (const float*)d_in[4];
  const float* b_time  = (const float*)d_in[5];
  const float* Wq1 = (const float*)d_in[6];  const float* bq1 = (const float*)d_in[7];
  const float* Wk1 = (const float*)d_in[8];  const float* bk1 = (const float*)d_in[9];
  const float* Wv1 = (const float*)d_in[10]; const float* bv1 = (const float*)d_in[11];
  const float* Ws1 = (const float*)d_in[12]; const float* bs1 = (const float*)d_in[13];
  const float* Wq2 = (const float*)d_in[14]; const float* bq2 = (const float*)d_in[15];
  const float* Wk2 = (const float*)d_in[16]; const float* bk2 = (const float*)d_in[17];
  const float* Wv2 = (const float*)d_in[18]; const float* bv2 = (const float*)d_in[19];
  const float* Ws2 = (const float*)d_in[20]; const float* bs2 = (const float*)d_in[21];
  const float* Wn  = (const float*)d_in[22]; const float* bn  = (const float*)d_in[23];
  const float* We1 = (const float*)d_in[24]; const float* be1 = (const float*)d_in[25];
  const float* We2 = (const float*)d_in[26]; const float* be2 = (const float*)d_in[27];
  float* out = (float*)d_out;

  // ---- workspace carve-up (256B aligned) ----
  size_t cur = 0;
  auto carve = [&](size_t bytes) -> char* {
    char* p = (char*)d_ws + cur;
    cur += (bytes + 255) & ~(size_t)255;
    return p;
  };
  float* d_cq1  = (float*)carve(256 * 4);
  float* d_ck1  = (float*)carve(256 * 4);
  float* d_cv1  = (float*)carve(256 * 4);
  float* d_cs1  = (float*)carve(64 * 4);
  float* d_q    = (float*)carve((size_t)NN * 256 * 4);
  float* d_k    = (float*)carve((size_t)NN * 256 * 4);
  float* d_v    = (float*)carve((size_t)NN * 256 * 4);
  float* d_skip = (float*)carve((size_t)NN * 64 * 4);
  float* d_lg   = (float*)carve((size_t)EE * 4 * 4);
  unsigned* d_nmax = (unsigned*)carve((size_t)NN * 4 * 4);
  float* d_den  = (float*)carve((size_t)NN * 4 * 4);
  float* d_agg  = (float*)carve((size_t)NN * 256 * 4);
  float* d_h1   = (float*)carve((size_t)NN * 64 * 4);
  float* d_h2   = (float*)carve((size_t)NN * 64 * 4);
  _Float16* d_h16 = (_Float16*)carve((size_t)NN * 64 * 2);
  _Float16* d_w16 = (_Float16*)carve((size_t)61440 * 2); // Wq2|Wk2|Wv2|Ws2|We1top|We1bot
  float* d_A    = (float*)carve((size_t)NN * 64 * 4);
  float* d_B    = (float*)carve((size_t)NN * 64 * 4);

  const int T = 256;
  auto blocks = [](size_t n, int t) { return (int)((n + t - 1) / t); };

  // 1. prep constants
  hipLaunchKernelGGL(k_prep, dim3(1), dim3(256), 0, stream,
                     tptr, W_time, b_time, Wq1, bq1, Wk1, bk1, Wv1, bv1, Ws1, bs1,
                     d_cq1, d_ck1, d_cv1, d_cs1);
  // 2. layer-1 projections
  hipLaunchKernelGGL(k_proj1, dim3(blocks((size_t)NN * 256, T)), dim3(T), 0, stream,
                     x, Wq1, Wk1, Wv1, d_cq1, d_ck1, d_cv1, d_q, d_k, d_v);
  hipLaunchKernelGGL(k_skip1, dim3(blocks((size_t)NN * 64, T)), dim3(T), 0, stream,
                     x, Ws1, d_cs1, d_skip);
  // 3. layer-1 attention
  hipLaunchKernelGGL(k_fill_u32, dim3(blocks(NN * 4, T)), dim3(T), 0, stream, d_nmax, NN * 4, NEG_INF_KEY);
  hipLaunchKernelGGL(k_fill_f32, dim3(blocks(NN * 4, T)), dim3(T), 0, stream, d_den, NN * 4, 0.f);
  hipLaunchKernelGGL(k_fill_f32, dim3(blocks((size_t)NN * 256, T)), dim3(T), 0, stream, d_agg, NN * 256, 0.f);
  hipLaunchKernelGGL(k_logits,   dim3(blocks((size_t)EE * 4, T)), dim3(T), 0, stream, d_q, d_k, ei, d_lg, d_nmax);
  hipLaunchKernelGGL(k_expdenom, dim3(blocks((size_t)EE * 4, T)), dim3(T), 0, stream, d_lg, ei, d_nmax, d_den);
  hipLaunchKernelGGL(k_scatter,  dim3(blocks((size_t)EE * 4, T)), dim3(T), 0, stream, d_v, d_lg, ei, d_den, d_agg);
  hipLaunchKernelGGL(k_combine,  dim3(blocks((size_t)NN * 64, T)), dim3(T), 0, stream, d_agg, d_skip, d_h1);

  // 4. f16 stage: h1 row-major; weights packed to WMMA lane order
  hipLaunchKernelGGL(k_f32_to_f16, dim3(blocks((size_t)NN * 64, T)), dim3(T), 0, stream, d_h1, d_h16, NN * 64);
  hipLaunchKernelGGL(k_pack_b, dim3(blocks(16384, T)), dim3(T), 0, stream, Wq2,        d_w16 + 0,     256);
  hipLaunchKernelGGL(k_pack_b, dim3(blocks(16384, T)), dim3(T), 0, stream, Wk2,        d_w16 + 16384, 256);
  hipLaunchKernelGGL(k_pack_b, dim3(blocks(16384, T)), dim3(T), 0, stream, Wv2,        d_w16 + 32768, 256);
  hipLaunchKernelGGL(k_pack_b, dim3(blocks(4096, T)),  dim3(T), 0, stream, Ws2,        d_w16 + 49152, 64);
  hipLaunchKernelGGL(k_pack_b, dim3(blocks(4096, T)),  dim3(T), 0, stream, We1,        d_w16 + 53248, 64);
  hipLaunchKernelGGL(k_pack_b, dim3(blocks(4096, T)),  dim3(T), 0, stream, We1 + 4096, d_w16 + 57344, 64);

  // 5. layer-2 projections via WMMA
  int gB256 = blocks((size_t)(NN / 16) * 16, 8);  // tiles / (8 waves per block)
  int gB64  = blocks((size_t)(NN / 16) * 4, 8);
  hipLaunchKernelGGL((k_gemm_wmma<256>), dim3(gB256), dim3(256), 0, stream, d_h16, d_w16 + 0,     bq2, d_q,    NN);
  hipLaunchKernelGGL((k_gemm_wmma<256>), dim3(gB256), dim3(256), 0, stream, d_h16, d_w16 + 16384, bk2, d_k,    NN);
  hipLaunchKernelGGL((k_gemm_wmma<256>), dim3(gB256), dim3(256), 0, stream, d_h16, d_w16 + 32768, bv2, d_v,    NN);
  hipLaunchKernelGGL((k_gemm_wmma<64>),  dim3(gB64),  dim3(256), 0, stream, d_h16, d_w16 + 49152, bs2, d_skip, NN);

  // 6. layer-2 attention
  hipLaunchKernelGGL(k_fill_u32, dim3(blocks(NN * 4, T)), dim3(T), 0, stream, d_nmax, NN * 4, NEG_INF_KEY);
  hipLaunchKernelGGL(k_fill_f32, dim3(blocks(NN * 4, T)), dim3(T), 0, stream, d_den, NN * 4, 0.f);
  hipLaunchKernelGGL(k_fill_f32, dim3(blocks((size_t)NN * 256, T)), dim3(T), 0, stream, d_agg, NN * 256, 0.f);
  hipLaunchKernelGGL(k_logits,   dim3(blocks((size_t)EE * 4, T)), dim3(T), 0, stream, d_q, d_k, ei, d_lg, d_nmax);
  hipLaunchKernelGGL(k_expdenom, dim3(blocks((size_t)EE * 4, T)), dim3(T), 0, stream, d_lg, ei, d_nmax, d_den);
  hipLaunchKernelGGL(k_scatter,  dim3(blocks((size_t)EE * 4, T)), dim3(T), 0, stream, d_v, d_lg, ei, d_den, d_agg);
  hipLaunchKernelGGL(k_combine,  dim3(blocks((size_t)NN * 64, T)), dim3(T), 0, stream, d_agg, d_skip, d_h2);

  // 7. node logits
  hipLaunchKernelGGL(k_node, dim3(blocks((size_t)NN * 4, T)), dim3(T), 0, stream, d_h2, Wn, bn, out);

  // 8. edge MLP precompute (A = h2@We1_top + b_e1, B = h2@We1_bot) via WMMA
  hipLaunchKernelGGL(k_f32_to_f16, dim3(blocks((size_t)NN * 64, T)), dim3(T), 0, stream, d_h2, d_h16, NN * 64);
  hipLaunchKernelGGL((k_gemm_wmma<64>), dim3(gB64), dim3(256), 0, stream, d_h16, d_w16 + 53248, be1,     d_A, NN);
  hipLaunchKernelGGL((k_gemm_wmma<64>), dim3(gB64), dim3(256), 0, stream, d_h16, d_w16 + 57344, nullptr, d_B, NN);

  // 9. fused pairwise edge logits
  hipLaunchKernelGGL(k_edge, dim3(BB * 4), dim3(256), 0, stream, d_A, d_B, We2, be2, out);
}
